// InteractionBlock_41300405518873
// MI455X (gfx1250) — compile-verified, matrix operand order checked by reading, not run on hardware
//
#include <hip/hip_runtime.h>
#include <hip/hip_bf16.h>

// ---------------------------------------------------------------------------
// SchNet InteractionBlock on gfx1250 (MI455X), WMMA f16-in/f32-acc everywhere.
// Edge MLP (~36 GFLOP) fused per 16-edge tile; [E,128] intermediates never hit
// HBM. h (25.6MB) and agg (25.6MB) stay L2-resident (192MB L2) so the edge
// gather (410MB of reads) and atomic scatter are served by L2, not HBM.
// HBM streaming ~280MB -> ~12us floor at 23.3 TB/s, balanced vs f16 WMMA.
// This revision stages the (pre-transposed, f16) weight matrices into LDS via
// the CDNA5 Tensor Data Mover (TENSOR_LOAD_TO_LDS + s_wait_tensorcnt) instead
// of VGPR-mediated copies, overlapping the DMA with per-wave tile staging.
// ---------------------------------------------------------------------------

#define HID 128
#define NF  128
#define NG  50
#define NGP 64          // NG padded to a multiple of 32 for K-steps
#define LOG2F_ 0.69314718055994531f
#define PI_OVER_CUTOFF 0.31415926535897931f  // pi / 10.0

typedef __attribute__((ext_vector_type(16))) _Float16 v16h;
typedef __attribute__((ext_vector_type(8)))  _Float16 v8h;
typedef __attribute__((ext_vector_type(4)))  _Float16 v4h;
typedef __attribute__((ext_vector_type(2)))  _Float16 v2h;
typedef __attribute__((ext_vector_type(8)))  float    v8f;
typedef __attribute__((ext_vector_type(4)))  unsigned int u32x4;
typedef __attribute__((ext_vector_type(8)))  int      i32x8;
typedef __attribute__((ext_vector_type(4)))  int      i32x4;

// Load one 16x32 f16 WMMA fragment from a row-major LDS tile.
// ISA layout (CDNA5 ISA): lanes 0-15 hold row M=lane with K blocks
// {0..7, 16..23}; lanes 16-31 hold row M=lane-16 with K {8..15, 24..31}.
// => two contiguous 8-half (16B) ds_load_b128 per lane.
__device__ __forceinline__ v16h load_frag(const _Float16* base, int stride, int lane) {
  const int row  = lane & 15;
  const int koff = (lane >> 4) << 3;           // 0 or 8
  const _Float16* p = base + row * stride + koff;
  v8h lo = *(const v8h*)(p);
  v8h hi = *(const v8h*)(p + 16);
  v16h f;
#pragma unroll
  for (int i = 0; i < 8; ++i) { f[i] = lo[i]; f[i + 8] = hi[i]; }
  return f;
}

__device__ __forceinline__ float sspf(float x) {
  // shifted softplus (fast transcendentals: v_exp_f32 / v_log_f32)
  return fmaxf(x, 0.f) + __logf(1.f + __expf(-fabsf(x))) - LOG2F_;
}

// ---------------------------------------------------------------------------
// TDM: DMA `nbytes` (multiple of 4) from contiguous global memory to LDS.
// D# per CDNA5 ISA: group0 = {count=1 | lds_addr |
// global_addr | type=2}; group1 = 1-D tensor, data_size=4B,
// tensor_dim0 = tile_dim0 = nbytes/4, tensor_dim0_stride = nbytes/4.
// Issue from one wave only; completion via s_wait_tensorcnt.
__device__ __forceinline__ void tdm_load_1d(unsigned lds_addr, const void* gsrc,
                                            unsigned nbytes) {
  const unsigned long long ga = (unsigned long long)(uintptr_t)gsrc;
  const unsigned ne = nbytes >> 2;                        // 4-byte elements
  u32x4 g0;
  g0[0] = 1u;                                             // count=1, user mode
  g0[1] = lds_addr;                                       // LDS byte address
  g0[2] = (unsigned)ga;                                   // global_addr[31:0]
  g0[3] = (unsigned)((ga >> 32) & 0x01FFFFFFu) | (2u << 30); // [56:32], type=2
  i32x8 g1;
  g1[0] = (int)(2u << 16);                                // data_size=2 (4B)
  g1[1] = (int)((ne & 0xFFFFu) << 16);                    // tensor_dim0[15:0]
  g1[2] = (int)(((ne >> 16) & 0xFFFFu) | (1u << 16));     // dim0[31:16], dim1=1
  g1[3] = (int)((ne & 0xFFFFu) << 16);                    // tile_dim0
  g1[4] = 0;                                              // tile_dim1/2 unused
  g1[5] = (int)ne;                                        // dim0_stride[31:0]
  g1[6] = 0;
  g1[7] = 0;
  const i32x4 z4 = {0, 0, 0, 0};
#if __has_include(<hip/amd_detail/amd_gfx1250_TDM.h>)
  const i32x8 z8 = {0, 0, 0, 0, 0, 0, 0, 0};              // therock: 6-arg form
  __builtin_amdgcn_tensor_load_to_lds(g0, g1, z4, z4, z8, 0);
#else
  __builtin_amdgcn_tensor_load_to_lds(g0, g1, z4, z4, 0); // ROCm 7.2: 5-arg
#endif
}

__device__ __forceinline__ unsigned lds_off(const void* p) {
  return (unsigned)(uintptr_t)p;   // generic addr low 32 bits = LDS byte offset
}

// ---------------------------------------------------------------------------
__global__ void zero_kernel(float4* __restrict__ p, size_t n4) {
  size_t i = (size_t)blockIdx.x * blockDim.x + threadIdx.x;
  if (i < n4) p[i] = make_float4(0.f, 0.f, 0.f, 0.f);
}

// ---------------------------------------------------------------------------
// One-shot: transpose + convert all weight matrices to f16 in workspace.
__global__ void prep_weights(const float* __restrict__ w1, const float* __restrict__ w2,
                             const float* __restrict__ l1w, const float* __restrict__ l2w,
                             const float* __restrict__ lw,
                             _Float16* __restrict__ w1t, _Float16* __restrict__ w2t,
                             _Float16* __restrict__ l1t, _Float16* __restrict__ l2t,
                             _Float16* __restrict__ lwt) {
  int i = blockIdx.x * blockDim.x + threadIdx.x;
  if (i < NF * NGP) {
    int n = i / NGP, k = i % NGP;
    w1t[i] = (k < NG) ? (_Float16)w1[k * NF + n] : (_Float16)0.f;
  }
  if (i < NF * NF) {
    int n = i >> 7, k = i & 127;
    w2t[i] = (_Float16)w2[k * NF + n];
    l1t[i] = (_Float16)l1w[k * NF + n];
    l2t[i] = (_Float16)l2w[k * NF + n];
    lwt[i] = (_Float16)lw[k * NF + n];
  }
}

// ---------------------------------------------------------------------------
// h = x @ lin1_w. 8 waves/block, one 16-row tile per wave.
__global__ void __launch_bounds__(256)
node_proj_kernel(const float* __restrict__ x, const _Float16* __restrict__ l1t,
                 float* __restrict__ h, int N) {
  __shared__ __align__(16) _Float16 Bt[NF][HID];
  __shared__ __align__(16) _Float16 Ash[8][16][HID];

  const int tid = threadIdx.x, lane = tid & 31, w = tid >> 5;

  if (w == 0) {  // TDM: weights -> LDS, no VGPR round-trip
    tdm_load_1d(lds_off(&Bt[0][0]), l1t, NF * HID * 2);
    __builtin_amdgcn_s_wait_tensorcnt(0);
  }
  asm volatile("" ::: "memory");
  const int r0 = blockIdx.x * 128 + w * 16;
  for (int t = lane; t < 16 * (HID / 4); t += 32) {   // float4 -> v4h
    int r = t >> 5, kq = t & 31;
    int row = r0 + r; row = row < N ? row : N - 1;
    float4 f = ((const float4*)(x + (size_t)row * HID))[kq];
    v4h hv = {(_Float16)f.x, (_Float16)f.y, (_Float16)f.z, (_Float16)f.w};
    *(v4h*)(&Ash[w][r][kq * 4]) = hv;
  }
  __syncthreads();

  const int nl = lane & 15;
  const int mh = (lane >> 4) << 3;
  const bool full = (r0 + 16) <= N;                   // wave-uniform
  float* pbase = h + (size_t)(r0 + mh) * NF + nl;     // one addr calc per wave

  v16h af[4];                                         // hoisted A fragments
#pragma unroll
  for (int kt = 0; kt < 4; ++kt)
    af[kt] = load_frag(&Ash[w][0][0] + kt * 32, HID, lane);

#pragma unroll
  for (int nt = 0; nt < 8; ++nt) {
    v8f acc = {};
#pragma unroll
    for (int kt = 0; kt < 4; ++kt) {
      v16h b = load_frag(&Bt[nt * 16][0] + kt * 32, HID, lane);
      acc = __builtin_amdgcn_wmma_f32_16x16x32_f16(false, af[kt], false, b,
                                                   (short)0, acc, false, false);
    }
    if (full) {
#pragma unroll
      for (int j = 0; j < 8; ++j) pbase[j * NF + nt * 16] = acc[j];  // imm offs
    } else {
#pragma unroll
      for (int j = 0; j < 8; ++j) {
        int row = r0 + mh + j;
        if (row < N) h[(size_t)row * NF + nt * 16 + nl] = acc[j];
      }
    }
  }
}

// ---------------------------------------------------------------------------
// Fused edge pipeline: W = ssp(ea@w1+b1)@w2+b2; W *= C; msg = h[src]*W;
// atomic scatter-add into agg[dst]. 16 edges/wave, 128 edges/block.
__global__ void __launch_bounds__(256)
edge_kernel(const long long* __restrict__ ei, const float* __restrict__ ew,
            const float* __restrict__ ea,
            const _Float16* __restrict__ w1t, const float* __restrict__ b1,
            const _Float16* __restrict__ w2t, const float* __restrict__ b2,
            const float* __restrict__ h, float* __restrict__ agg, int E) {
  __shared__ __align__(16) _Float16 B1t[NF][NGP];
  __shared__ __align__(16) _Float16 B2t[NF][NF];
  __shared__ __align__(16) _Float16 A1[8][16][NGP];
  __shared__ __align__(16) _Float16 Ys[8][16][NF];
  __shared__ float b1s[NF], b2s[NF];
  __shared__ float cenv[8][16];
  __shared__ int   srcs[8][16], dsts[8][16];

  const int tid = threadIdx.x, lane = tid & 31, w = tid >> 5;

  if (w == 0) {  // TDM: both weight matrices -> LDS
    tdm_load_1d(lds_off(&B1t[0][0]), w1t, NF * NGP * 2);
    tdm_load_1d(lds_off(&B2t[0][0]), w2t, NF * NF * 2);
    __builtin_amdgcn_s_wait_tensorcnt(0);
  }
  asm volatile("" ::: "memory");
  if (tid < NF) { b1s[tid] = b1[tid]; b2s[tid] = b2[tid]; }

  const int e0 = blockIdx.x * 128 + w * 16;
  if (lane < 16) {
    int e  = e0 + lane;
    int ec = e < E ? e : E - 1;
    srcs[w][lane] = (int)ei[ec];
    dsts[w][lane] = (int)ei[(size_t)E + ec];
    float c = 0.5f * (__cosf(ew[ec] * PI_OVER_CUTOFF) + 1.f);
    cenv[w][lane] = (e < E) ? c : 0.f;   // OOB edges contribute exactly 0
  }
  for (int t = lane; t < 16 * (NGP / 2); t += 32) {   // float2 -> v2h (8B align)
    int r = t >> 5, kp = t & 31;
    int e = e0 + r; e = e < E ? e : E - 1;
    float2 f = make_float2(0.f, 0.f);
    if (kp < NG / 2) f = ((const float2*)(ea + (size_t)e * NG))[kp];
    v2h hv = {(_Float16)f.x, (_Float16)f.y};
    *(v2h*)(&A1[w][r][kp * 2]) = hv;
  }
  __syncthreads();

  const int nl = lane & 15;
  const int mh = (lane >> 4) << 3;

  // GEMM1: [16x64]@[64x128] + b1, ssp -> Ys (f16)
  {
    v16h af[2];
#pragma unroll
    for (int kt = 0; kt < 2; ++kt)
      af[kt] = load_frag(&A1[w][0][0] + kt * 32, NGP, lane);
#pragma unroll
    for (int nt = 0; nt < 8; ++nt) {
      const int n = nt * 16 + nl;
      const float bb = b1s[n];
      v8f acc;
#pragma unroll
      for (int j = 0; j < 8; ++j) acc[j] = bb;
#pragma unroll
      for (int kt = 0; kt < 2; ++kt) {
        v16h b = load_frag(&B1t[nt * 16][0] + kt * 32, NGP, lane);
        acc = __builtin_amdgcn_wmma_f32_16x16x32_f16(false, af[kt], false, b,
                                                     (short)0, acc, false, false);
      }
      _Float16* yp = &Ys[w][mh][n];                  // imm offsets j*256B
#pragma unroll
      for (int j = 0; j < 8; ++j) yp[j * NF] = (_Float16)sspf(acc[j]);
    }
  }

  // Hoist per-lane gather/scatter bases (8 rows each): inner loop uses
  // immediate offsets only (nt*64 bytes).
  const float* ph[8]; float* pd[8]; float cv[8];
#pragma unroll
  for (int j = 0; j < 8; ++j) {
    int m = mh + j;
    ph[j] = h   + (size_t)srcs[w][m] * NF + nl;
    pd[j] = agg + (size_t)dsts[w][m] * NF + nl;
    cv[j] = cenv[w][m];
  }

  // GEMM2: [16x128]@[128x128] + b2, * C[row] * h[src][n] -> atomic agg[dst][n]
  {
    v16h af[4];
#pragma unroll
    for (int kt = 0; kt < 4; ++kt)
      af[kt] = load_frag(&Ys[w][0][0] + kt * 32, NF, lane);
#pragma unroll
    for (int nt = 0; nt < 8; ++nt) {
      const int n = nt * 16 + nl;
      const float bb = b2s[n];
      v8f acc;
#pragma unroll
      for (int j = 0; j < 8; ++j) acc[j] = bb;
#pragma unroll
      for (int kt = 0; kt < 4; ++kt) {
        v16h b = load_frag(&B2t[nt * 16][0] + kt * 32, NF, lane);
        acc = __builtin_amdgcn_wmma_f32_16x16x32_f16(false, af[kt], false, b,
                                                     (short)0, acc, false, false);
      }
#pragma unroll
      for (int j = 0; j < 8; ++j) {
        float v = acc[j] * cv[j] * ph[j][nt * 16];
        atomicAdd(pd[j] + nt * 16, v);
      }
    }
  }
}

// ---------------------------------------------------------------------------
// out = ssp(agg @ lin2_w + lin2_b) @ lin_w + lin_b
__global__ void __launch_bounds__(256)
out_kernel(const float* __restrict__ agg,
           const _Float16* __restrict__ l2t, const float* __restrict__ lin2_b,
           const _Float16* __restrict__ lwt, const float* __restrict__ lin_b,
           float* __restrict__ out, int N) {
  __shared__ __align__(16) _Float16 Bt2[HID][NF];
  __shared__ __align__(16) _Float16 BtL[HID][HID];
  __shared__ __align__(16) _Float16 Ash[8][16][NF];
  __shared__ __align__(16) _Float16 Ys[8][16][HID];
  __shared__ float b2s[HID], bLs[HID];

  const int tid = threadIdx.x, lane = tid & 31, w = tid >> 5;

  if (w == 0) {  // TDM: both weight matrices -> LDS
    tdm_load_1d(lds_off(&Bt2[0][0]), l2t, HID * NF * 2);
    tdm_load_1d(lds_off(&BtL[0][0]), lwt, HID * HID * 2);
    __builtin_amdgcn_s_wait_tensorcnt(0);
  }
  asm volatile("" ::: "memory");
  if (tid < HID) { b2s[tid] = lin2_b[tid]; bLs[tid] = lin_b[tid]; }

  const int r0 = blockIdx.x * 128 + w * 16;
  for (int t = lane; t < 16 * (NF / 4); t += 32) {    // float4 -> v4h
    int r = t >> 5, kq = t & 31;
    int row = r0 + r; row = row < N ? row : N - 1;
    float4 f = ((const float4*)(agg + (size_t)row * NF))[kq];
    v4h hv = {(_Float16)f.x, (_Float16)f.y, (_Float16)f.z, (_Float16)f.w};
    *(v4h*)(&Ash[w][r][kq * 4]) = hv;
  }
  __syncthreads();

  const int nl = lane & 15;
  const int mh = (lane >> 4) << 3;

  // GEMM1 + bias + ssp -> Ys
  {
    v16h af[4];
#pragma unroll
    for (int kt = 0; kt < 4; ++kt)
      af[kt] = load_frag(&Ash[w][0][0] + kt * 32, NF, lane);
#pragma unroll
    for (int nt = 0; nt < 8; ++nt) {
      const int n = nt * 16 + nl;
      const float bb = b2s[n];
      v8f acc;
#pragma unroll
      for (int j = 0; j < 8; ++j) acc[j] = bb;
#pragma unroll
      for (int kt = 0; kt < 4; ++kt) {
        v16h b = load_frag(&Bt2[nt * 16][0] + kt * 32, NF, lane);
        acc = __builtin_amdgcn_wmma_f32_16x16x32_f16(false, af[kt], false, b,
                                                     (short)0, acc, false, false);
      }
      _Float16* yp = &Ys[w][mh][n];
#pragma unroll
      for (int j = 0; j < 8; ++j) yp[j * NF] = (_Float16)sspf(acc[j]);
    }
  }

  const bool full = (r0 + 16) <= N;                   // wave-uniform
  float* pbase = out + (size_t)(r0 + mh) * HID + nl;

  // GEMM2 + bias -> out
  {
    v16h af[4];
#pragma unroll
    for (int kt = 0; kt < 4; ++kt)
      af[kt] = load_frag(&Ys[w][0][0] + kt * 32, HID, lane);
#pragma unroll
    for (int nt = 0; nt < 8; ++nt) {
      const int n = nt * 16 + nl;
      const float bb = bLs[n];
      v8f acc;
#pragma unroll
      for (int j = 0; j < 8; ++j) acc[j] = bb;
#pragma unroll
      for (int kt = 0; kt < 4; ++kt) {
        v16h b = load_frag(&BtL[nt * 16][0] + kt * 32, HID, lane);
        acc = __builtin_amdgcn_wmma_f32_16x16x32_f16(false, af[kt], false, b,
                                                     (short)0, acc, false, false);
      }
      if (full) {
#pragma unroll
        for (int j = 0; j < 8; ++j) pbase[j * HID + nt * 16] = acc[j];
      } else {
#pragma unroll
        for (int j = 0; j < 8; ++j) {
          int row = r0 + mh + j;
          if (row < N) out[(size_t)row * HID + nt * 16 + nl] = acc[j];
        }
      }
    }
  }
}

// ---------------------------------------------------------------------------
extern "C" void kernel_launch(void* const* d_in, const int* in_sizes, int n_in,
                              void* d_out, int out_size, void* d_ws, size_t ws_size,
                              hipStream_t stream) {
  const float*     x   = (const float*)d_in[0];
  const long long* ei  = (const long long*)d_in[1];   // int64 [2,E]
  const float*     ew  = (const float*)d_in[2];
  const float*     ea  = (const float*)d_in[3];
  const float*     w1  = (const float*)d_in[4];
  const float*     b1  = (const float*)d_in[5];
  const float*     w2  = (const float*)d_in[6];
  const float*     b2  = (const float*)d_in[7];
  const float*     l1w = (const float*)d_in[8];
  const float*     l2w = (const float*)d_in[9];
  const float*     l2b = (const float*)d_in[10];
  const float*     lw  = (const float*)d_in[11];
  const float*     lb  = (const float*)d_in[12];

  const int N = in_sizes[0] / HID;
  const int E = in_sizes[2];

  float* h    = (float*)d_ws;               // [N,128] f32 projected nodes
  float* agg  = h + (size_t)N * NF;         // [N,128] f32 scatter accumulator
  _Float16* w1t = (_Float16*)(agg + (size_t)N * NF);  // f16 transposed weights
  _Float16* w2t = w1t + NF * NGP;
  _Float16* l1t = w2t + NF * NF;
  _Float16* l2t = l1t + NF * NF;
  _Float16* lwt = l2t + NF * NF;
  float* outf = (float*)d_out;

  prep_weights<<<(NF * NF + 255) / 256, 256, 0, stream>>>(w1, w2, l1w, l2w, lw,
                                                          w1t, w2t, l1t, l2t, lwt);
  const size_t nz4 = (size_t)N * NF / 4;
  zero_kernel<<<(unsigned)((nz4 + 255) / 256), 256, 0, stream>>>((float4*)agg, nz4);
  node_proj_kernel<<<(N + 127) / 128, 256, 0, stream>>>(x, l1t, h, N);
  edge_kernel<<<(E + 127) / 128, 256, 0, stream>>>(ei, ew, ea, w1t, b1, w2t, b2,
                                                   h, agg, E);
  out_kernel<<<(N + 127) / 128, 256, 0, stream>>>(agg, l2t, l2b, lwt, lb, outf, N);
}
